// ConvFFNMs_79216376807579
// MI455X (gfx1250) — matrix operand order repaired and checked by reading, not compiled, not run on hardware
//
#include <hip/hip_runtime.h>
#include <hip/hip_bf16.h>
#include <stdint.h>

typedef __attribute__((ext_vector_type(16))) _Float16 v16h;
typedef __attribute__((ext_vector_type(8)))  _Float16 v8h;
typedef __attribute__((ext_vector_type(4)))  _Float16 v4h;
typedef __attribute__((ext_vector_type(8)))  float    v8f;

static constexpr int T_DIM = 4;
static constexpr int B_DIM = 8;
static constexpr int C_DIM = 256;
static constexpr int V_DIM = 512;
static constexpr int H_DIM = 1024;

__device__ __forceinline__ void wait_asynccnt0() {
#if __has_builtin(__builtin_amdgcn_s_wait_asynccnt)
    __builtin_amdgcn_s_wait_asynccnt(0);
#else
    asm volatile("s_wait_asynccnt 0" ::: "memory");
#endif
}

// ---------------------------------------------------------------------------
// prep: convert weights to f16, fold b1+BN2 into per-row scale/bias for GEMM1,
// identity scale + b2 bias for GEMM2.
// ---------------------------------------------------------------------------
__global__ void __launch_bounds__(256)
prep_kernel(const float* __restrict__ W1, const float* __restrict__ W2,
            const float* __restrict__ b1,
            const float* __restrict__ g2, const float* __restrict__ be2,
            const float* __restrict__ mu2, const float* __restrict__ va2,
            const float* __restrict__ b2,
            _Float16* __restrict__ w1h, _Float16* __restrict__ w2h,
            float* __restrict__ rs1, float* __restrict__ rb1,
            float* __restrict__ rs2, float* __restrict__ rb2)
{
    const int i = blockIdx.x * blockDim.x + threadIdx.x;
    if (i < H_DIM * C_DIM) {
        w1h[i] = (_Float16)W1[i];
        w2h[i] = (_Float16)W2[i];
    }
    if (i < H_DIM) {
        const float s = g2[i] * rsqrtf(va2[i] + 1e-5f);
        rs1[i] = s;
        rb1[i] = (b1[i] - mu2[i]) * s + be2[i];
    }
    if (i < C_DIM) {
        rs2[i] = 1.0f;
        rb2[i] = b2[i];
    }
}

// ---------------------------------------------------------------------------
// LIF1: BN1 inline + LIF recurrence over (t, v) per (b, c) lane. f16 spikes.
// ---------------------------------------------------------------------------
__global__ void __launch_bounds__(256)
lif1_kernel(const float* __restrict__ x,
            const float* __restrict__ g1, const float* __restrict__ be1,
            const float* __restrict__ mu1, const float* __restrict__ va1,
            _Float16* __restrict__ s1)
{
    const int idx = blockIdx.x * blockDim.x + threadIdx.x;   // [0, B*C)
    const int b = idx / C_DIM;
    const int c = idx % C_DIM;
    const float inv = g1[c] * rsqrtf(va1[c] + 1e-5f);
    const float sh  = be1[c] - mu1[c] * inv;
    float vmem = 0.0f;
    for (int t = 0; t < T_DIM; ++t) {
        const size_t base = (((size_t)t * B_DIM + b) * C_DIM + c) * V_DIM;
        const float4* xp = (const float4*)(x + base);
        v4h* sp = (v4h*)(s1 + base);
        for (int v4 = 0; v4 < V_DIM / 4; ++v4) {
            const float4 xv = xp[v4];
            const float xa[4] = { xv.x, xv.y, xv.z, xv.w };
            v4h o;
            #pragma unroll
            for (int j = 0; j < 4; ++j) {
                const float h = xa[j] * inv + sh;      // BN1
                vmem += (h - vmem) * 0.5f;             // decay_input, tau=2
                const float s = (vmem >= 1.0f) ? 1.0f : 0.0f;
                vmem *= (1.0f - s);                    // hard reset
                o[j] = (_Float16)s;
            }
            sp[v4] = o;
        }
    }
}

// ---------------------------------------------------------------------------
// LIF2: input already BN2-normalized (fused into GEMM1 epilogue).
// ---------------------------------------------------------------------------
__global__ void __launch_bounds__(256)
lif2_kernel(const float* __restrict__ h2, _Float16* __restrict__ s2)
{
    const int idx = blockIdx.x * blockDim.x + threadIdx.x;   // [0, B*H)
    const int b  = idx / H_DIM;
    const int hc = idx % H_DIM;
    float vmem = 0.0f;
    for (int t = 0; t < T_DIM; ++t) {
        const size_t base = (((size_t)t * B_DIM + b) * H_DIM + hc) * V_DIM;
        const float4* xp = (const float4*)(h2 + base);
        v4h* sp = (v4h*)(s2 + base);
        for (int v4 = 0; v4 < V_DIM / 4; ++v4) {
            const float4 xv = xp[v4];
            const float xa[4] = { xv.x, xv.y, xv.z, xv.w };
            v4h o;
            #pragma unroll
            for (int j = 0; j < 4; ++j) {
                vmem += (xa[j] - vmem) * 0.5f;
                const float s = (vmem >= 1.0f) ? 1.0f : 0.0f;
                vmem *= (1.0f - s);
                o[j] = (_Float16)s;
            }
            sp[v4] = o;
        }
    }
}

// ---------------------------------------------------------------------------
// Batched WMMA GEMM: Out[z,m,n] = (sum_k A[m,k]*B[z,k,n]) * scale[m] + bias[m]
// Block tile 64x128, K-step 64 (2 WMMA sub-steps), 128 threads = 4 waves 2x2;
// wave tile 32x64 -> 16 WMMAs per staged tile, B fragments shared by M-tiles.
// Double-buffered LDS. A staged via global_load_async_to_lds_b128 (ASYNCcnt);
// B staged through registers with transpose scatter so fragment reads are
// contiguous 16B at stride 72 halves (conflict-free). All fragments for a
// sub-step are preloaded before the WMMA block so dscnt waits stagger.
// ---------------------------------------------------------------------------
__global__ void __launch_bounds__(128)
wmma_gemm_kernel(const _Float16* __restrict__ A,
                 const _Float16* __restrict__ Bmat,
                 const float* __restrict__ rowscale,
                 const float* __restrict__ rowbias,
                 float* __restrict__ Out,
                 int M, int N, int K)
{
    constexpr int BM = 64, BN = 128, BK = 64, AS = BK + 8;   // stride 72 halves
    constexpr int A_HALVES = BM * AS;                        // 4608
    constexpr int B_HALVES = BN * AS;                        // 9216
    __shared__ _Float16 smem[2 * A_HALVES + 2 * B_HALVES];   // 55296 B
    const unsigned A_OFF[2] = { 0u, (unsigned)A_HALVES };
    const unsigned B_OFF[2] = { 2u * A_HALVES, 2u * A_HALVES + B_HALVES };

    const int tid  = threadIdx.x;
    const int lane = tid & 31;
    const int wv   = tid >> 5;
    const int wm   = wv & 1;          // wave M half
    const int wn   = wv >> 1;         // wave N half
    const int z    = blockIdx.z;
    const int nblk = blockIdx.x * BN;
    const int mblk = blockIdx.y * BM;

    const _Float16* __restrict__ Bz = Bmat + (size_t)z * K * N;
    const uint64_t Abase = (uint64_t)(uintptr_t)A;

    // ---- staging helpers (uniform control flow) ----
    auto stageA_async = [&](int buf, int kk) {
        const unsigned ldsb = (unsigned)(uintptr_t)(&smem[A_OFF[buf]]);
        #pragma unroll
        for (int i = 0; i < 4; ++i) {
            const int chunk = tid + i * 128;          // 512 chunks of 8 halves
            const int m  = chunk >> 3;
            const int k0 = (chunk & 7) << 3;
            const unsigned lds = ldsb + (unsigned)(m * AS + k0) * 2u;
            const unsigned gof = (unsigned)(((mblk + m) * K + kk + k0) * 2);
            asm volatile("global_load_async_to_lds_b128 %0, %1, %2"
                         :: "v"(lds), "v"(gof), "s"(Abase) : "memory");
        }
    };
    auto loadB_regs = [&](v8h* breg, int kk) {
        #pragma unroll
        for (int i = 0; i < 8; ++i) {
            const int chunk = tid + i * 128;          // 1024 chunks of 8 halves
            const int k  = chunk >> 4;
            const int n0 = (chunk & 15) << 3;
            breg[i] = *(const v8h*)(Bz + (size_t)(kk + k) * N + (nblk + n0));
        }
    };
    auto storeB_lds = [&](const v8h* breg, int buf) {
        _Float16* Bb = &smem[B_OFF[buf]];
        #pragma unroll
        for (int i = 0; i < 8; ++i) {
            const int chunk = tid + i * 128;
            const int k  = chunk >> 4;
            const int n0 = (chunk & 15) << 3;
            #pragma unroll
            for (int j = 0; j < 8; ++j) Bb[(n0 + j) * AS + k] = breg[i][j];
        }
    };

    // ---- prologue: stage k-step 0 into buffer 0 ----
    {
        v8h breg[8];
        loadB_regs(breg, 0);
        stageA_async(0, 0);
        storeB_lds(breg, 0);
        wait_asynccnt0();
    }

    v8f acc[2][4] = {};
    const int lr = lane & 15;
    const int hi = lane >> 4;
    const int ak = hi << 3;           // A frag k-offset: 0 / 8
    const int bk = hi << 3;           // B frag k-offset: 0 / 8
    const int nsteps = K / BK;

    union U16 { v16h v; v8h h[2]; };

    for (int s = 0; s < nsteps; ++s) {
        __syncthreads();
        const int cur = s & 1;
        const int nxt = cur ^ 1;
        const bool more = (s + 1) < nsteps;

        v8h breg[8];
        if (more) {
            loadB_regs(breg, (s + 1) * BK);           // VMEM in flight
            stageA_async(nxt, (s + 1) * BK);          // ASYNC in flight
        }

        const _Float16* Ab = &smem[A_OFF[cur]];
        const _Float16* Bb = &smem[B_OFF[cur]];
        #pragma unroll
        for (int ks = 0; ks < BK / 32; ++ks) {
            const int ko = ks * 32;
            // preload ALL fragments for this sub-step (12 ds_load_b128),
            // then issue 8 WMMAs -> dscnt waits can stagger.
            U16 a0, a1, bf[4];
            a0.h[0] = *(const v8h*)(Ab + (wm * 32 + lr) * AS + ko + ak);
            a0.h[1] = *(const v8h*)(Ab + (wm * 32 + lr) * AS + ko + ak + 16);
            a1.h[0] = *(const v8h*)(Ab + (wm * 32 + 16 + lr) * AS + ko + ak);
            a1.h[1] = *(const v8h*)(Ab + (wm * 32 + 16 + lr) * AS + ko + ak + 16);
            #pragma unroll
            for (int j = 0; j < 4; ++j) {
                const _Float16* Bp = Bb + (wn * 64 + j * 16 + lr) * AS + ko;
                bf[j].h[0] = *(const v8h*)(Bp + bk);
                bf[j].h[1] = *(const v8h*)(Bp + bk + 16);
            }
            #pragma unroll
            for (int j = 0; j < 4; ++j) {
                acc[0][j] = __builtin_amdgcn_wmma_f32_16x16x32_f16(
                    false, a0.v, false, bf[j].v, (short)0, acc[0][j], false, false);
                acc[1][j] = __builtin_amdgcn_wmma_f32_16x16x32_f16(
                    false, a1.v, false, bf[j].v, (short)0, acc[1][j], false, false);
            }
        }

        if (more) {
            storeB_lds(breg, nxt);    // waits loadcnt, overlaps with WMMA issue
            wait_asynccnt0();         // A tile for buffer nxt landed
        }
    }

    // ---- epilogue: fused affine (BN2+b1 for GEMM1, +b2 for GEMM2) ----
    const int col = nblk + wn * 64 + lr;
    #pragma unroll
    for (int mt = 0; mt < 2; ++mt) {
        const int mrow = mblk + wm * 32 + mt * 16 + (hi << 3);
        #pragma unroll
        for (int q = 0; q < 8; ++q) {
            const int m = mrow + q;
            const float sc = rowscale[m];
            const float bi = rowbias[m];
            #pragma unroll
            for (int j = 0; j < 4; ++j) {
                Out[((size_t)z * M + m) * N + (col + j * 16)] = acc[mt][j][q] * sc + bi;
            }
        }
    }
}

// ---------------------------------------------------------------------------
extern "C" void kernel_launch(void* const* d_in, const int* in_sizes, int n_in,
                              void* d_out, int out_size, void* d_ws, size_t ws_size,
                              hipStream_t stream) {
    (void)in_sizes; (void)n_in; (void)out_size; (void)ws_size;
    const float* x   = (const float*)d_in[0];
    const float* g1  = (const float*)d_in[1];
    const float* be1 = (const float*)d_in[2];
    const float* mu1 = (const float*)d_in[3];
    const float* va1 = (const float*)d_in[4];
    const float* W1  = (const float*)d_in[5];
    const float* b1  = (const float*)d_in[6];
    const float* g2  = (const float*)d_in[7];
    const float* be2 = (const float*)d_in[8];
    const float* mu2 = (const float*)d_in[9];
    const float* va2 = (const float*)d_in[10];
    const float* W2  = (const float*)d_in[11];
    const float* b2  = (const float*)d_in[12];
    float* out = (float*)d_out;

    char* ws = (char*)d_ws;
    size_t o = 0;
    _Float16* w1h = (_Float16*)(ws + o); o += (size_t)H_DIM * C_DIM * 2;       // 512 KB
    _Float16* w2h = (_Float16*)(ws + o); o += (size_t)C_DIM * H_DIM * 2;       // 512 KB
    float* rs1 = (float*)(ws + o); o += (size_t)H_DIM * 4;
    float* rb1 = (float*)(ws + o); o += (size_t)H_DIM * 4;
    float* rs2 = (float*)(ws + o); o += (size_t)C_DIM * 4;
    float* rb2 = (float*)(ws + o); o += (size_t)C_DIM * 4;
    _Float16* s1 = (_Float16*)(ws + o); o += (size_t)T_DIM*B_DIM*C_DIM*V_DIM * 2;  // 8 MB
    float*    h2 = (float*)   (ws + o); o += (size_t)T_DIM*B_DIM*H_DIM*V_DIM * 4;  // 67 MB
    _Float16* s2 = (_Float16*)(ws + o); o += (size_t)T_DIM*B_DIM*H_DIM*V_DIM * 2;  // 33.5 MB

    prep_kernel<<<(H_DIM * C_DIM) / 256, 256, 0, stream>>>(
        W1, W2, b1, g2, be2, mu2, va2, b2, w1h, w2h, rs1, rb1, rs2, rb2);

    lif1_kernel<<<(B_DIM * C_DIM) / 256, 256, 0, stream>>>(
        x, g1, be1, mu1, va1, s1);

    wmma_gemm_kernel<<<dim3(V_DIM / 128, H_DIM / 64, T_DIM * B_DIM), 128, 0, stream>>>(
        w1h, s1, rs1, rb1, h2, H_DIM, V_DIM, C_DIM);

    lif2_kernel<<<(B_DIM * H_DIM) / 256, 256, 0, stream>>>(h2, s2);

    wmma_gemm_kernel<<<dim3(V_DIM / 128, C_DIM / 64, T_DIM * B_DIM), 128, 0, stream>>>(
        w2h, s2, rs2, rb2, out, C_DIM, V_DIM, H_DIM);
}
